// VectorQuantizedVAE_64871186038982
// MI455X (gfx1250) — matrix operand order repaired and checked by reading, not compiled, not run on hardware
//
#include <hip/hip_runtime.h>

typedef __attribute__((ext_vector_type(16))) __bf16 v16bf;
typedef __attribute__((ext_vector_type(8)))  __bf16 bf16x8;
typedef __attribute__((ext_vector_type(4)))  __bf16 bf16x4;
typedef __attribute__((ext_vector_type(8)))  float  v8f;
typedef __attribute__((ext_vector_type(4)))  int    v4i;

#define B_ROWS 16384
#define IN_DIM 1024
#define HID    2048
#define KCODES 8192
#define DDIM   128

#if __has_builtin(__builtin_amdgcn_global_load_async_to_lds_b128) && \
    __has_builtin(__builtin_amdgcn_s_wait_asynccnt)
#define USE_ASYNC_LDS 1
#else
#define USE_ASYNC_LDS 0
#endif

#if USE_ASYNC_LDS
#define ASYNC_B128(gp, lp)                                                     \
  __builtin_amdgcn_global_load_async_to_lds_b128(                              \
      (__attribute__((address_space(1))) v4i*)(gp),                            \
      (__attribute__((address_space(3))) v4i*)(lp), 0, 0)
#endif

static __device__ __forceinline__ v16bf frag16(bf16x8 lo, bf16x8 hi) {
  return __builtin_shufflevector(lo, hi, 0,1,2,3,4,5,6,7,8,9,10,11,12,13,14,15);
}

// ---------------- conversion kernels ----------------

__global__ void f32_to_bf16_vec4(const float* __restrict__ s, __bf16* __restrict__ d, int n4) {
  int i = blockIdx.x * blockDim.x + threadIdx.x;
  int stride = gridDim.x * blockDim.x;
  for (; i < n4; i += stride) {
    float4 v = ((const float4*)s)[i];
    bf16x4 o = { (__bf16)v.x, (__bf16)v.y, (__bf16)v.z, (__bf16)v.w };
    ((bf16x4*)d)[i] = o;
  }
}

// src [R][C] f32 -> dst [C][R] bf16 (transpose), R,C multiples of 32
__global__ void transpose_f32_to_bf16(const float* __restrict__ src, __bf16* __restrict__ dst,
                                      int R, int C) {
  __shared__ float tile[32][33];
  int c0 = blockIdx.x * 32, r0 = blockIdx.y * 32;
  int tx = threadIdx.x, ty = threadIdx.y;   // 32 x 8
#pragma unroll
  for (int j = 0; j < 4; ++j)
    tile[ty + 8*j][tx] = src[(size_t)(r0 + ty + 8*j) * C + (c0 + tx)];
  __syncthreads();
#pragma unroll
  for (int j = 0; j < 4; ++j)
    dst[(size_t)(c0 + ty + 8*j) * R + (r0 + tx)] = (__bf16)tile[tx][ty + 8*j];
}

__global__ void codebook_norms(const float* __restrict__ cb, float* __restrict__ cnorm) {
  int row  = blockIdx.x * 8 + (threadIdx.x >> 5);
  int lane = threadIdx.x & 31;
  float4 v = ((const float4*)(cb + (size_t)row * DDIM))[lane];  // 32 lanes * 4 = 128
  float s = v.x*v.x + v.y*v.y + v.z*v.z + v.w*v.w;
#pragma unroll
  for (int off = 16; off >= 1; off >>= 1) s += __shfl_xor(s, off, 32);
  if (lane == 0) cnorm[row] = s;
}

// ---------------- WMMA GEMM: C = act(A @ B^T_stored + bias) ----------------
// A: [M][K] bf16 row-major. BT: [N][K] bf16 row-major (pre-transposed weights).
// Block tile 128x128, BK=64; 8 waves in 4(M) x 2(N); wave tile 32x64 = 2x4 WMMA.
// ACT 0: relu -> bf16 out   ACT 1: sigmoid -> f32 out   ACT 2: none -> f32 + bf16 out
template <int ACT>
__global__ __launch_bounds__(256)
void gemm_wmma_bf16(const __bf16* __restrict__ A, const __bf16* __restrict__ BT,
                    const float* __restrict__ bias,
                    float* __restrict__ outF, __bf16* __restrict__ outB,
                    int M, int N, int K) {
  __shared__ __align__(16) __bf16 As[128 * 72];   // 128 rows x BK=64, stride 72 (144B, conflict-free)
  __shared__ __align__(16) __bf16 Bs[128 * 72];   // 128 rows x BK=64
  const int tid  = threadIdx.x;
  const int lane = tid & 31;
  const int wave = tid >> 5;
  const int bm = blockIdx.x * 128, bn = blockIdx.y * 128;
  const int wm = (wave & 3) * 32, wn = (wave >> 2) * 64;
  const int hf = lane >> 4, l = lane & 15;

  v8f acc[2][4] = {};

  // tile loaders: 128 rows, 2 threads/row, 4x16B chunks each
  const int row  = tid >> 1;
  const int cb4  = (tid & 1) * 4;
  const __bf16* ag = A  + (size_t)(bm + row) * K + cb4 * 8;
  const __bf16* bg = BT + (size_t)(bn + row) * K + cb4 * 8;

  for (int k0 = 0; k0 < K; k0 += 64) {
    __syncthreads();
#if USE_ASYNC_LDS
#pragma unroll
    for (int j = 0; j < 4; ++j) {
      ASYNC_B128(ag + k0 + j*8, &As[row*72 + (cb4+j)*8]);
      ASYNC_B128(bg + k0 + j*8, &Bs[row*72 + (cb4+j)*8]);
    }
    __builtin_amdgcn_s_wait_asynccnt(0);
#else
#pragma unroll
    for (int j = 0; j < 4; ++j) {
      *(uint4*)(&As[row*72 + (cb4+j)*8]) = *(const uint4*)(ag + k0 + j*8);
      *(uint4*)(&Bs[row*72 + (cb4+j)*8]) = *(const uint4*)(bg + k0 + j*8);
    }
#endif
    if (k0 + 64 < K) {  // global_prefetch_b8 of the next K tile
      __builtin_prefetch(ag + k0 + 64, 0, 1);
      __builtin_prefetch(bg + k0 + 64, 0, 1);
    }
    __syncthreads();
#pragma unroll
    for (int ks = 0; ks < 2; ++ks) {
      const int kk = ks * 32;
      v16bf afr[2], bfr[4];
#pragma unroll
      for (int im = 0; im < 2; ++im) {
        // A layout: lanes 0-15 (hf=0): K = kk+{0..7, 16..23}; lanes 16-31: +8
        const __bf16* p = &As[(wm + im*16 + l)*72 + kk + hf*8];
        afr[im] = frag16(*(const bf16x8*)p, *(const bf16x8*)(p + 16));
      }
#pragma unroll
      for (int in = 0; in < 4; ++in) {
        // B layout: lanes 0-15 (hf=0): K = kk+{0..7, 16..23}; lanes 16-31: +8
        const __bf16* p = &Bs[(wn + in*16 + l)*72 + kk + hf*8];
        bfr[in] = frag16(*(const bf16x8*)p, *(const bf16x8*)(p + 16));
      }
#pragma unroll
      for (int im = 0; im < 2; ++im)
#pragma unroll
        for (int in = 0; in < 4; ++in)
          acc[im][in] = __builtin_amdgcn_wmma_f32_16x16x32_bf16(
              false, afr[im], false, bfr[in], (short)0, acc[im][in], false, false);
    }
  }
  // epilogue: D layout — VGPR r: lanes 0-15 -> (M=r, N=lane), lanes 16-31 -> (M=r+8)
#pragma unroll
  for (int in = 0; in < 4; ++in) {
    const int col = bn + wn + in*16 + l;
    const float bv = bias[col];
#pragma unroll
    for (int im = 0; im < 2; ++im) {
      const int rbase = bm + wm + im*16 + hf*8;
#pragma unroll
      for (int r = 0; r < 8; ++r) {
        float v = acc[im][in][r] + bv;
        const size_t o = (size_t)(rbase + r) * N + col;
        if (ACT == 0)      { v = v > 0.f ? v : 0.f; outB[o] = (__bf16)v; }
        else if (ACT == 1) { v = 1.0f / (1.0f + __expf(-v)); outF[o] = v; }
        else               { outF[o] = v; outB[o] = (__bf16)v; }
      }
    }
  }
}

// ---------------- VQ argmin: per row, argmin_k (||c_k||^2 - 2 z.c_k) ----------------
__global__ __launch_bounds__(256)
void vq_argmin(const __bf16* __restrict__ zb, const __bf16* __restrict__ cb,
               const float* __restrict__ cnorm, int* __restrict__ idxOut) {
  __shared__ float cns[KCODES];
  for (int i = threadIdx.x; i < KCODES; i += 256) cns[i] = cnorm[i];
  __syncthreads();
  const int lane = threadIdx.x & 31, wave = threadIdx.x >> 5;
  const int hf = lane >> 4, l = lane & 15;
  const int m0 = (blockIdx.x * 8 + wave) * 16;

  v16bf afr[4];                 // z rows m0..m0+15, D=128 = 4 K-steps of 32
#pragma unroll
  for (int c = 0; c < 4; ++c) {
    const __bf16* p = zb + (size_t)(m0 + l) * DDIM + 32*c + hf*8;
    afr[c] = frag16(*(const bf16x8*)p, *(const bf16x8*)(p + 16));
  }
  float minv[8]; int mini[8];
#pragma unroll
  for (int r = 0; r < 8; ++r) { minv[r] = 3.0e38f; mini[r] = 0; }

  for (int n0 = 0; n0 < KCODES; n0 += 16) {
    v8f acc = {};
#pragma unroll
    for (int c = 0; c < 4; ++c) {
      // B = C^T: WMMA-N = code id, WMMA-K = d. Lane holds d = 32c+hf*8+{0..7, 16..23}.
      const __bf16* p = cb + (size_t)(n0 + l) * DDIM + 32*c + hf*8;
      v16bf bfr = frag16(*(const bf16x8*)p, *(const bf16x8*)(p + 16));
      acc = __builtin_amdgcn_wmma_f32_16x16x32_bf16(false, afr[c], false, bfr,
                                                    (short)0, acc, false, false);
    }
    const float cn = cns[n0 + l];
    const int n = n0 + l;
#pragma unroll
    for (int r = 0; r < 8; ++r) {
      const float dv = cn - 2.0f * acc[r];
      if (dv < minv[r]) { minv[r] = dv; mini[r] = n; }
    }
  }
  // reduce across 16 lanes of each half (masks 1..8 stay within a half)
#pragma unroll
  for (int r = 0; r < 8; ++r) {
    float v = minv[r]; int i = mini[r];
#pragma unroll
    for (int off = 8; off >= 1; off >>= 1) {
      const float ov = __shfl_xor(v, off, 32);
      const int   oi = __shfl_xor(i, off, 32);
      if (ov < v || (ov == v && oi < i)) { v = ov; i = oi; }
    }
    if (l == 0) idxOut[m0 + hf*8 + r] = i;
  }
}

// ---------------- gather z_q, MSE partials, emit indices-as-float ----------------
__global__ void vq_gather_mse(const int* __restrict__ idx, const float* __restrict__ cb,
                              const float* __restrict__ z, __bf16* __restrict__ zqb,
                              float* __restrict__ outIdxF, float* __restrict__ mseAcc) {
  const int lane = threadIdx.x & 31, wave = threadIdx.x >> 5;
  const int row = blockIdx.x * 8 + wave;
  const int id = idx[row];
  const int d0 = lane * 4;
  float4 q  = *(const float4*)(cb + (size_t)id  * DDIM + d0);
  float4 zz = *(const float4*)(z  + (size_t)row * DDIM + d0);
  const float dx = q.x - zz.x, dy = q.y - zz.y, dz = q.z - zz.z, dw = q.w - zz.w;
  float s = dx*dx + dy*dy + dz*dz + dw*dw;
  bf16x4 o = { (__bf16)q.x, (__bf16)q.y, (__bf16)q.z, (__bf16)q.w };
  *(bf16x4*)(zqb + (size_t)row * DDIM + d0) = o;
#pragma unroll
  for (int off = 16; off >= 1; off >>= 1) s += __shfl_xor(s, off, 32);
  if (lane == 0) {
    atomicAdd(mseAcc, s);
    outIdxF[row] = (float)id;
  }
}

__global__ void zero_one(float* p) { if (threadIdx.x == 0 && blockIdx.x == 0) p[0] = 0.f; }

__global__ void finalize_loss(const float* __restrict__ mseAcc, float* __restrict__ lossOut) {
  if (threadIdx.x == 0) lossOut[0] = 1.25f * (mseAcc[0] / (float)(B_ROWS * DDIM));
}

// ---------------- launcher ----------------
extern "C" void kernel_launch(void* const* d_in, const int* in_sizes, int n_in,
                              void* d_out, int out_size, void* d_ws, size_t ws_size,
                              hipStream_t stream) {
  (void)in_sizes; (void)n_in; (void)out_size; (void)ws_size;
  const float* x  = (const float*)d_in[0];
  const float* W1 = (const float*)d_in[1];
  const float* b1 = (const float*)d_in[2];
  const float* W2 = (const float*)d_in[3];
  const float* b2 = (const float*)d_in[4];
  const float* cb = (const float*)d_in[5];
  const float* W3 = (const float*)d_in[6];
  const float* b3 = (const float*)d_in[7];
  const float* W4 = (const float*)d_in[8];
  const float* b4 = (const float*)d_in[9];
  float* out = (float*)d_out;

  char* ws = (char*)d_ws;
  size_t off = 0;
  auto alloc = [&](size_t bytes) -> void* {
    void* p = ws + off;
    off = (off + bytes + 255) & ~(size_t)255;
    return p;
  };
  __bf16* x_b   = (__bf16*)alloc((size_t)B_ROWS * IN_DIM * 2);
  __bf16* w1t   = (__bf16*)alloc((size_t)HID * IN_DIM * 2);   // W1^T [HID][IN_DIM]
  __bf16* w2t   = (__bf16*)alloc((size_t)DDIM * HID * 2);     // W2^T [DDIM][HID]
  __bf16* w3t   = (__bf16*)alloc((size_t)HID * DDIM * 2);     // W3^T [HID][DDIM]
  __bf16* w4t   = (__bf16*)alloc((size_t)IN_DIM * HID * 2);   // W4^T [IN_DIM][HID]
  __bf16* cb_b  = (__bf16*)alloc((size_t)KCODES * DDIM * 2);
  float*  cnorm = (float*) alloc((size_t)KCODES * 4);
  __bf16* h_b   = (__bf16*)alloc((size_t)B_ROWS * HID * 2);   // reused for h and h2
  float*  z_f   = (float*) alloc((size_t)B_ROWS * DDIM * 4);
  __bf16* z_b   = (__bf16*)alloc((size_t)B_ROWS * DDIM * 2);
  __bf16* zq_b  = (__bf16*)alloc((size_t)B_ROWS * DDIM * 2);
  int*    idx   = (int*)   alloc((size_t)B_ROWS * 4);
  float*  mse   = (float*) alloc(256);

  float* recon   = out;                                  // [B_ROWS * IN_DIM]
  float* lossOut = out + (size_t)B_ROWS * IN_DIM;        // [1]
  float* idxOut  = lossOut + 1;                          // [B_ROWS]

  // one-time precision conversion + weight transposes
  f32_to_bf16_vec4<<<1024, 256, 0, stream>>>(x,  x_b,  B_ROWS * IN_DIM / 4);
  f32_to_bf16_vec4<<<512,  256, 0, stream>>>(cb, cb_b, KCODES * DDIM / 4);
  transpose_f32_to_bf16<<<dim3(HID/32,    IN_DIM/32), dim3(32,8), 0, stream>>>(W1, w1t, IN_DIM, HID);
  transpose_f32_to_bf16<<<dim3(DDIM/32,   HID/32),    dim3(32,8), 0, stream>>>(W2, w2t, HID, DDIM);
  transpose_f32_to_bf16<<<dim3(HID/32,    DDIM/32),   dim3(32,8), 0, stream>>>(W3, w3t, DDIM, HID);
  transpose_f32_to_bf16<<<dim3(IN_DIM/32, HID/32),    dim3(32,8), 0, stream>>>(W4, w4t, HID, IN_DIM);
  codebook_norms<<<KCODES/8, 256, 0, stream>>>(cb, cnorm);

  // encoder: h = relu(x@W1+b1); z = h@W2+b2 (f32 + bf16)
  gemm_wmma_bf16<0><<<dim3(B_ROWS/128, HID/128),  256, 0, stream>>>(x_b, w1t, b1, nullptr, h_b, B_ROWS, HID,  IN_DIM);
  gemm_wmma_bf16<2><<<dim3(B_ROWS/128, DDIM/128), 256, 0, stream>>>(h_b, w2t, b2, z_f,     z_b, B_ROWS, DDIM, HID);

  // vector quantization
  vq_argmin<<<B_ROWS/(16*8), 256, 0, stream>>>(z_b, cb_b, cnorm, idx);
  zero_one<<<1, 64, 0, stream>>>(mse);
  vq_gather_mse<<<B_ROWS/8, 256, 0, stream>>>(idx, cb, z_f, zq_b, idxOut, mse);
  finalize_loss<<<1, 32, 0, stream>>>(mse, lossOut);

  // decoder: h2 = relu(zq@W3+b3); recon = sigmoid(h2@W4+b4)
  gemm_wmma_bf16<0><<<dim3(B_ROWS/128, HID/128),    256, 0, stream>>>(zq_b, w3t, b3, nullptr, h_b, B_ROWS, HID,    DDIM);
  gemm_wmma_bf16<1><<<dim3(B_ROWS/128, IN_DIM/128), 256, 0, stream>>>(h_b,  w4t, b4, recon,  nullptr, B_ROWS, IN_DIM, HID);
}
